// Transformer_79637283602930
// MI455X (gfx1250) — matrix operand order repaired and checked by reading, not compiled, not run on hardware
//
#include <hip/hip_runtime.h>

// ---------------------------------------------------------------------------
// MI455X (gfx1250) transformer forward. Wave32, WMMA bf16 path.
//   B=64, T=256, F=255, D=256, H=8, DK=DV=64, FF=1024, L=4, OUT=1
// GEMMs: v_wmma_f32_16x16x32_bf16, K-step 64, LDS tiles in fragment-friendly
// layouts (contiguous v16bf loads -> ds_load_b128 x2 per fragment).
// sched_group_barrier pins "all DS reads, then all WMMAs" per iteration so
// the waitcnt pass emits counted s_wait_dscnt (DS overlaps matrix issue).
// ---------------------------------------------------------------------------

typedef __attribute__((ext_vector_type(16))) __bf16 v16bf;
typedef __attribute__((ext_vector_type(8)))  float  v8f;

#if __has_builtin(__builtin_amdgcn_sched_group_barrier)
#define SCHED_GROUP(mask, size, id) __builtin_amdgcn_sched_group_barrier(mask, size, id)
#else
#define SCHED_GROUP(mask, size, id)
#endif

static __device__ __forceinline__ __bf16 f2bf(float f) {
  unsigned u = __builtin_bit_cast(unsigned, f) + 0x8000u;  // round half up
  unsigned short s = (unsigned short)(u >> 16);
  return __builtin_bit_cast(__bf16, s);
}

// pack two floats into two bf16 (lo = a, hi = b)
static __device__ __forceinline__ unsigned pack2(float a, float b) {
  unsigned ua = __builtin_bit_cast(unsigned, a) + 0x8000u;
  unsigned ub = __builtin_bit_cast(unsigned, b) + 0x8000u;
  return (ua >> 16) | (ub & 0xFFFF0000u);
}

// swap bits 3 and 4 of a K index (within a 32-chunk): makes each half-wave's
// A-fragment elements contiguous (half0: K 0-7,16-23 ; half1: K 8-15,24-31).
static __device__ __forceinline__ int swap34(int k) {
  return (k & ~24) | ((k & 8) << 1) | ((k & 16) >> 1);
}
// permute within a 64-wide K row: preserve bit5 chunk, swap34 inside chunk
static __device__ __forceinline__ int perm64(int k) {
  return (k & 32) | swap34(k & 31);
}

// ---------------------------------------------------------------------------
// Time2Vec: h[b,t,0:255] = wb[t]*x + bb[t]; h[b,t,255] = sin(x_row . wa + ba[t])
// ---------------------------------------------------------------------------
__global__ void __launch_bounds__(256)
k_time2vec(const float* __restrict__ x, const float* __restrict__ wb,
           const float* __restrict__ bb, const float* __restrict__ wa,
           const float* __restrict__ ba, float* __restrict__ h) {
  __shared__ float red[256];
  const int row = blockIdx.x;          // b*256 + t
  const int t   = row & 255;
  const int j   = threadIdx.x;
  const float* xr = x + (long)row * 255;
  float p = (j < 255) ? xr[j] * wa[j] : 0.f;
  red[j] = p;
  __syncthreads();
  for (int off = 128; off; off >>= 1) {
    if (j < off) red[j] += red[j + off];
    __syncthreads();
  }
  float dp = red[0] + ba[t];
  float* hr = h + (long)row * 256;
  if (j < 255) hr[j] = wb[t] * xr[j] + bb[t];
  if (j == 0)  hr[255] = __sinf(dp);
}

// ---------------------------------------------------------------------------
// Generic WMMA GEMM:  C = act(A[M,K] * B[K,N] + bias (+residual))
// Block tile 64x64, 4 waves, each wave a 16(M)x64(N) strip, K-step 64.
// LDS layouts:
//   At[rr*64 + kk]              -> A frag k2: contiguous 16 @ row*64+k2*32+hl*16
//   Btc[cc*64 + kr] (col-major) -> B frag k2: contiguous 16 @ col*64+k2*32+hl*16
// Requires: grid covers M,N exactly (M%64==0, N%64==0) and K%64==0.
// z-batching: B += z*bStride, C += z*cStride, bias += z*biasStride.
// Output row remap: crow = (row/rpc)*crs + row%rpc  (for [B,H,T,*] layouts).
// ---------------------------------------------------------------------------
__global__ void __launch_bounds__(128)
k_gemm(const float* __restrict__ A, int lda,
       const float* __restrict__ Bm, int ldb, int bStride,
       float* __restrict__ C, int ldc, int cStride,
       const float* __restrict__ bias, int biasStride,
       const float* __restrict__ res,
       int M, int N, int K, int rpc, int crs, int relu) {
  __shared__ __bf16 At[64 * 64];
  __shared__ __bf16 Btc[64 * 64];
  const int tid  = threadIdx.x;
  const int w    = tid >> 5;
  const int lane = tid & 31;
  const int hl   = lane >> 4;      // half-wave
  const int r    = lane & 15;
  const int m0   = blockIdx.y * 64;
  const int n0   = blockIdx.x * 64;
  const int z    = blockIdx.z;
  Bm += (long)z * bStride;
  C  += (long)z * cStride;
  const float* bz = bias ? (bias + (long)z * biasStride) : nullptr;

  v8f acc[4];
  v8f zer = {};
  acc[0] = zer; acc[1] = zer; acc[2] = zer; acc[3] = zer;

  // Staging coordinates: invariant across the K loop (p & 31 == tid & 31 etc.)
  const int rr0  = tid >> 5;                 // A row step base (rows rr0+4i)
  const int kkA  = (tid & 31) * 2;           // even K within tile
  const int posA = rr0 * 64 + kkA;
  const float* aP = A + (long)(m0 + rr0) * lda + kkA;
  const int ccB  = tid & 63;                 // B column
  const int kr0  = (tid >> 6) * 2;           // K pair base (kr0+4i)
  const int posB = ccB * 64 + kr0;
  const float* bP = Bm + (long)kr0 * ldb + n0 + ccB;

  for (int k0 = 0; k0 < K; k0 += 64) {
    __syncthreads();
    // Stage A tile 64x64 (f32x2 loads -> packed b32 DS stores)
#pragma unroll
    for (int i = 0; i < 16; ++i) {
      float2 va = *(const float2*)(aP + (long)(i * 4) * lda + k0);
      *(unsigned*)&At[posA + i * 256] = pack2(va.x, va.y);
    }
    // Stage B tile 64x64 column-major (pairs along K)
#pragma unroll
    for (int i = 0; i < 16; ++i) {
      const float* bpi = bP + (long)(k0 + i * 4) * ldb;
      *(unsigned*)&Btc[posB + i * 4] = pack2(bpi[0], bpi[ldb]);
    }
    if (k0 + 64 < K) {   // prefetch next K tile -> global_prefetch_b8
      __builtin_prefetch(aP + (long)(k0 + 64), 0, 1);
      __builtin_prefetch(bP + (long)(k0 + 64) * ldb, 0, 1);
    }
    __syncthreads();

    // Load all fragments first, then run the WMMA chain
    v16bf af[2], bfv[2][4];
#pragma unroll
    for (int k2 = 0; k2 < 2; ++k2) {
      af[k2] = *(const v16bf*)&At[(w * 16 + r) * 64 + k2 * 32 + hl * 16];
#pragma unroll
      for (int n = 0; n < 4; ++n)
        bfv[k2][n] = *(const v16bf*)&Btc[(n * 16 + r) * 64 + k2 * 32 + hl * 16];
    }
#pragma unroll
    for (int k2 = 0; k2 < 2; ++k2)
#pragma unroll
      for (int n = 0; n < 4; ++n)
        acc[n] = __builtin_amdgcn_wmma_f32_16x16x32_bf16(
            false, af[k2], false, bfv[k2][n], (short)0, acc[n], false, false);
    // Pin schedule: 20 DS reads (10 v16bf frags x 2 b128), then 8 WMMAs.
    SCHED_GROUP(0x100, 20, 0);
    SCHED_GROUP(0x008, 8, 0);
  }

  // Epilogue: C layout M=e+8*hl, N=lane%16 (grids cover M,N exactly)
#pragma unroll
  for (int n = 0; n < 4; ++n) {
#pragma unroll
    for (int e = 0; e < 8; ++e) {
      int row = m0 + w * 16 + e + hl * 8;
      int col = n0 + n * 16 + r;
      float val = acc[n][e];
      if (bz)  val += bz[col];
      if (res) val += res[(long)row * N + col];
      if (relu) val = fmaxf(val, 0.f);
      int crow = (row / rpc) * crs + (row % rpc);
      C[(long)crow * ldc + col] = val;
    }
  }
}

// ---------------------------------------------------------------------------
// Fused attention: per (b,h), 64-query tile, online softmax, scale 1/sqrt(T).
// Q,K,V: [B,H,T,64] f32.  O: [B,T,H*64] f32.
// LDS: Kt[s][k] (k-contiguous) for QK^T B-frags; Vtt[dv][s] (s-contiguous)
// for PV B-frags; Pt per-wave [q][s] (s-contiguous, no permute) for PV A-frags.
// ---------------------------------------------------------------------------
__global__ void __launch_bounds__(128)
k_attn(const float* __restrict__ Q, const float* __restrict__ Kg,
       const float* __restrict__ Vg, float* __restrict__ O) {
  __shared__ __bf16 Kt[64 * 64];
  __shared__ __bf16 Vtt[64 * 64];
  __shared__ __bf16 Pt[4][16 * 64];
  const int tid  = threadIdx.x;
  const int w    = tid >> 5;
  const int lane = tid & 31;
  const int hl   = lane >> 4;
  const int r    = lane & 15;
  const int b    = blockIdx.z, hh = blockIdx.y;
  const int qbase = blockIdx.x * 64;
  const long base = (long)(b * 8 + hh) * 256 * 64;

  // Q fragments (16 rows x 64 features = two 16x32 bf16 frags) via float4 loads
  const float* qrow = Q + base + (long)(qbase + w * 16 + r) * 64;
  v16bf qa[2];
#pragma unroll
  for (int k2 = 0; k2 < 2; ++k2) {
    float4 q0 = *(const float4*)(qrow + k2 * 32 + hl * 16);
    float4 q1 = *(const float4*)(qrow + k2 * 32 + hl * 16 + 4);
    float4 q2 = *(const float4*)(qrow + k2 * 32 + hl * 16 + 8);
    float4 q3 = *(const float4*)(qrow + k2 * 32 + hl * 16 + 12);
    qa[k2][0]  = f2bf(q0.x); qa[k2][1]  = f2bf(q0.y);
    qa[k2][2]  = f2bf(q0.z); qa[k2][3]  = f2bf(q0.w);
    qa[k2][4]  = f2bf(q1.x); qa[k2][5]  = f2bf(q1.y);
    qa[k2][6]  = f2bf(q1.z); qa[k2][7]  = f2bf(q1.w);
    qa[k2][8]  = f2bf(q2.x); qa[k2][9]  = f2bf(q2.y);
    qa[k2][10] = f2bf(q2.z); qa[k2][11] = f2bf(q2.w);
    qa[k2][12] = f2bf(q3.x); qa[k2][13] = f2bf(q3.y);
    qa[k2][14] = f2bf(q3.z); qa[k2][15] = f2bf(q3.w);
  }

  v8f oacc[4];
  v8f zer = {};
  oacc[0] = zer; oacc[1] = zer; oacc[2] = zer; oacc[3] = zer;
  float rmax[8], rsum[8];
#pragma unroll
  for (int e = 0; e < 8; ++e) { rmax[e] = -1e30f; rsum[e] = 0.f; }

  // staging coordinates (invariant)
  const int rrK = tid >> 5;              // K rows rrK + 4i
  const int kpK = (tid & 31) * 2;        // even feature
  const int dvV = tid & 63;              // V column (dv)
  const int spV = (tid >> 6) * 2;        // even s base (spV + 4i)

  for (int s0 = 0; s0 < 256; s0 += 64) {
    __syncthreads();
    // stage K (k-pairs, packed b32) and V transposed (s-pairs, packed b32)
#pragma unroll
    for (int i = 0; i < 16; ++i) {
      int rr = rrK + i * 4;
      float2 kv = *(const float2*)(Kg + base + (long)(s0 + rr) * 64 + kpK);
      *(unsigned*)&Kt[rr * 64 + kpK] = pack2(kv.x, kv.y);
      int sp = spV + i * 4;
      float v0 = Vg[base + (long)(s0 + sp) * 64 + dvV];
      float v1 = Vg[base + (long)(s0 + sp + 1) * 64 + dvV];
      *(unsigned*)&Vtt[dvV * 64 + sp] = pack2(v0, v1);
    }
    __syncthreads();

    // scores = Q * K^T * (1/sqrt(T)),  T = 256 -> scale 1/16
    v16bf kf[2][4];
#pragma unroll
    for (int k2 = 0; k2 < 2; ++k2)
#pragma unroll
      for (int n = 0; n < 4; ++n)
        kf[k2][n] = *(const v16bf*)&Kt[(n * 16 + r) * 64 + k2 * 32 + hl * 16];
    v8f sc[4];
#pragma unroll
    for (int n = 0; n < 4; ++n) {
      v8f a = zer;
      a = __builtin_amdgcn_wmma_f32_16x16x32_bf16(
          false, qa[0], false, kf[0][n], (short)0, a, false, false);
      a = __builtin_amdgcn_wmma_f32_16x16x32_bf16(
          false, qa[1], false, kf[1][n], (short)0, a, false, false);
#pragma unroll
      for (int e = 0; e < 8; ++e) a[e] *= 0.0625f;
      sc[n] = a;
    }
    // Pin schedule: 16 DS reads (8 frags), then 8 WMMAs.
    SCHED_GROUP(0x100, 16, 1);
    SCHED_GROUP(0x008, 8, 1);

    // online softmax: row = e + 8*hl, columns striped over 16 lanes
    float tmax[8];
#pragma unroll
    for (int e = 0; e < 8; ++e)
      tmax[e] = fmaxf(fmaxf(sc[0][e], sc[1][e]), fmaxf(sc[2][e], sc[3][e]));
#pragma unroll
    for (int m = 8; m; m >>= 1)
#pragma unroll
      for (int e = 0; e < 8; ++e)
        tmax[e] = fmaxf(tmax[e], __shfl_xor(tmax[e], m, 16));

    float alpha[8], tsum[8];
#pragma unroll
    for (int e = 0; e < 8; ++e) {
      float nm = fmaxf(rmax[e], tmax[e]);
      alpha[e] = __expf(rmax[e] - nm);
      rmax[e] = nm;
      tsum[e] = 0.f;
    }
#pragma unroll
    for (int n = 0; n < 4; ++n) {
      int pcol = n * 16 + r;
#pragma unroll
      for (int e = 0; e < 8; ++e) {
        float p = __expf(sc[n][e] - rmax[e]);
        tsum[e] += p;
        Pt[w][(e + hl * 8) * 64 + pcol] = f2bf(p);
      }
    }
#pragma unroll
    for (int m = 8; m; m >>= 1)
#pragma unroll
      for (int e = 0; e < 8; ++e) tsum[e] += __shfl_xor(tsum[e], m, 16);
#pragma unroll
    for (int e = 0; e < 8; ++e) rsum[e] = rsum[e] * alpha[e] + tsum[e];
#pragma unroll
    for (int n = 0; n < 4; ++n)
#pragma unroll
      for (int e = 0; e < 8; ++e) oacc[n][e] *= alpha[e];

    // O += P * V : load all fragments, then WMMA chain
    v16bf pa[2], vf[2][4];
#pragma unroll
    for (int k2 = 0; k2 < 2; ++k2) {
      pa[k2] = *(const v16bf*)&Pt[w][r * 64 + k2 * 32 + hl * 16];
#pragma unroll
      for (int n = 0; n < 4; ++n)
        vf[k2][n] = *(const v16bf*)&Vtt[(n * 16 + r) * 64 + k2 * 32 + hl * 16];
    }
#pragma unroll
    for (int k2 = 0; k2 < 2; ++k2)
#pragma unroll
      for (int n = 0; n < 4; ++n)
        oacc[n] = __builtin_amdgcn_wmma_f32_16x16x32_bf16(
            false, pa[k2], false, vf[k2][n], (short)0, oacc[n], false, false);
    // Pin schedule: 20 DS reads (10 frags), then 8 WMMAs.
    SCHED_GROUP(0x100, 20, 2);
    SCHED_GROUP(0x008, 8, 2);
  }

  // normalize and write O[b, t, h*64 + dv]
#pragma unroll
  for (int n = 0; n < 4; ++n)
#pragma unroll
    for (int e = 0; e < 8; ++e) {
      int t = qbase + w * 16 + e + hl * 8;
      O[(long)(b * 256 + t) * 512 + hh * 64 + n * 16 + r] =
          oacc[n][e] / rsum[e];
    }
}

// ---------------------------------------------------------------------------
// LayerNorm over D=256, one block per row.
// ---------------------------------------------------------------------------
__global__ void __launch_bounds__(256)
k_layernorm(const float* __restrict__ X, const float* __restrict__ g,
            const float* __restrict__ be, float* __restrict__ Y) {
  __shared__ float red[256];
  const int row = blockIdx.x;
  const int j = threadIdx.x;
  float v = X[(long)row * 256 + j];
  red[j] = v;
  __syncthreads();
  for (int off = 128; off; off >>= 1) {
    if (j < off) red[j] += red[j + off];
    __syncthreads();
  }
  float mu = red[0] * (1.f / 256.f);
  __syncthreads();
  float d = v - mu;
  red[j] = d * d;
  __syncthreads();
  for (int off = 128; off; off >>= 1) {
    if (j < off) red[j] += red[j + off];
    __syncthreads();
  }
  float var = red[0] * (1.f / 256.f);
  Y[(long)row * 256 + j] = g[j] * d * rsqrtf(var + 1e-6f) + be[j];
}

// ---------------------------------------------------------------------------
// Final head: out[b] = f2[b,:] . wo_f + bo_f   ([64,64] x [64,1])
// ---------------------------------------------------------------------------
__global__ void __launch_bounds__(64)
k_final(const float* __restrict__ f2, const float* __restrict__ wof,
        const float* __restrict__ bof, float* __restrict__ out) {
  int b = threadIdx.x;
  float s = 0.f;
#pragma unroll 8
  for (int j = 0; j < 64; ++j) s += f2[b * 64 + j] * wof[j];
  out[b] = s + bof[0];
}

// ---------------------------------------------------------------------------
// Launch. Inputs in setup_inputs() dict order:
// 0:x 1:t2v_wb 2:t2v_bb 3:t2v_wa 4:t2v_ba
// 5+16l: wq bq wk bk wv bv wo bo g1 be1 wf1 bf1 wf2 bf2 g2 be2 (l=0..3)
// 69:wd1 70:bd1 71:wd2 72:bd2 73:wo_f 74:bo_f
// ---------------------------------------------------------------------------
extern "C" void kernel_launch(void* const* d_in, const int* in_sizes, int n_in,
                              void* d_out, int out_size, void* d_ws, size_t ws_size,
                              hipStream_t stream) {
  (void)in_sizes; (void)n_in; (void)out_size; (void)ws_size;
  const float* x   = (const float*)d_in[0];
  const float* twb = (const float*)d_in[1];
  const float* tbb = (const float*)d_in[2];
  const float* twa = (const float*)d_in[3];
  const float* tba = (const float*)d_in[4];

  // Workspace (f32): h 4.19M | q/k/v 8.39M each | o 8.39M | t1 16.78M | t2 4.19M
  float* h  = (float*)d_ws;
  float* qb = h  + 4194304;
  float* kb = qb + 8388608;
  float* vb = kb + 8388608;
  float* ob = vb + 8388608;
  float* t1 = ob + 8388608;
  float* t2 = t1 + 16777216;

  const int M = 16384;  // B*T

  k_time2vec<<<16384, 256, 0, stream>>>(x, twb, tbb, twa, tba, h);

  for (int l = 0; l < 4; ++l) {
    const float* wq  = (const float*)d_in[5 + l * 16 + 0];
    const float* bq  = (const float*)d_in[5 + l * 16 + 1];
    const float* wk  = (const float*)d_in[5 + l * 16 + 2];
    const float* bk  = (const float*)d_in[5 + l * 16 + 3];
    const float* wv  = (const float*)d_in[5 + l * 16 + 4];
    const float* bv  = (const float*)d_in[5 + l * 16 + 5];
    const float* wo  = (const float*)d_in[5 + l * 16 + 6];
    const float* bo  = (const float*)d_in[5 + l * 16 + 7];
    const float* g1  = (const float*)d_in[5 + l * 16 + 8];
    const float* be1 = (const float*)d_in[5 + l * 16 + 9];
    const float* wf1 = (const float*)d_in[5 + l * 16 + 10];
    const float* bf1 = (const float*)d_in[5 + l * 16 + 11];
    const float* wf2 = (const float*)d_in[5 + l * 16 + 12];
    const float* bf2 = (const float*)d_in[5 + l * 16 + 13];
    const float* g2  = (const float*)d_in[5 + l * 16 + 14];
    const float* be2 = (const float*)d_in[5 + l * 16 + 15];

    // Per-head QKV projections: [B*T,256] x [256,64] -> [B,H,T,64]
    k_gemm<<<dim3(1, 256, 8), 128, 0, stream>>>(
        h, 256, wq, 64, 16384, qb, 64, 16384, bq, 64, nullptr,
        M, 64, 256, 256, 2048, 0);
    k_gemm<<<dim3(1, 256, 8), 128, 0, stream>>>(
        h, 256, wk, 64, 16384, kb, 64, 16384, bk, 64, nullptr,
        M, 64, 256, 256, 2048, 0);
    k_gemm<<<dim3(1, 256, 8), 128, 0, stream>>>(
        h, 256, wv, 64, 16384, vb, 64, 16384, bv, 64, nullptr,
        M, 64, 256, 256, 2048, 0);

    k_attn<<<dim3(4, 8, 64), 128, 0, stream>>>(qb, kb, vb, ob);

    // o @ wo + bo + residual(h) -> t2 ; LN -> h
    k_gemm<<<dim3(4, 256, 1), 128, 0, stream>>>(
        ob, 512, wo, 256, 0, t2, 256, 0, bo, 0, h,
        M, 256, 512, M, 0, 0);
    k_layernorm<<<16384, 256, 0, stream>>>(t2, g1, be1, h);

    // FFN
    k_gemm<<<dim3(16, 256, 1), 128, 0, stream>>>(
        h, 256, wf1, 1024, 0, t1, 1024, 0, bf1, 0, nullptr,
        M, 1024, 256, M, 0, 1);
    k_gemm<<<dim3(4, 256, 1), 128, 0, stream>>>(
        t1, 1024, wf2, 256, 0, t2, 256, 0, bf2, 0, h,
        M, 256, 1024, M, 0, 0);
    k_layernorm<<<16384, 256, 0, stream>>>(t2, g2, be2, h);
  }

  const float* wd1 = (const float*)d_in[69];
  const float* bd1 = (const float*)d_in[70];
  const float* wd2 = (const float*)d_in[71];
  const float* bd2 = (const float*)d_in[72];
  const float* wof = (const float*)d_in[73];
  const float* bof = (const float*)d_in[74];

  // h viewed as [64, 65536]
  k_gemm<<<dim3(8, 1, 1), 128, 0, stream>>>(
      h, 65536, wd1, 512, 0, t1, 512, 0, bd1, 0, nullptr,
      64, 512, 65536, 64, 0, 1);
  k_gemm<<<dim3(1, 1, 1), 128, 0, stream>>>(
      t1, 512, wd2, 64, 0, t2, 64, 0, bd2, 0, nullptr,
      64, 64, 512, 64, 0, 1);
  k_final<<<1, 64, 0, stream>>>(t2, wof, bof, (float*)d_out);
}
